// DeepSeekMoEFFN_33011118637694
// MI455X (gfx1250) — matrix operand order repaired and checked by reading, not compiled, not run on hardware
//
#include <hip/hip_runtime.h>
#include <math.h>

#define N_TOK 16384
#define DIM   1024
#define NEXP  8
#define HALFD 1536
#define MTILE 32                 // tokens per block (2 x 16-row WMMA subtiles)
#define HS_STRIDE (HALFD + 8)    // 1544 bf16 -> 3088B rows, banks spread, 16B aligned

typedef __attribute__((ext_vector_type(16))) __bf16   v16bf;
typedef __attribute__((ext_vector_type(8)))  float    v8f;
typedef __attribute__((ext_vector_type(4)))  float    f32x4;
typedef __attribute__((ext_vector_type(4)))  unsigned int u32x4;

union Frag {
    v16bf v;
    u32x4 q[2];
    unsigned short s[16];
};

__device__ __forceinline__ unsigned short bf16_bits(float f) {
    union { __bf16 b; unsigned short u; } c;
    c.b = (__bf16)f;            // lowers to v_cvt_pk_bf16_f32 / fma_mix*_bf16
    return c.u;
}

__device__ __forceinline__ float fast_rcp(float f) {
#if __has_builtin(__builtin_amdgcn_rcpf)
    return __builtin_amdgcn_rcpf(f);     // v_rcp_f32
#else
    return 1.0f / f;
#endif
}

__device__ __forceinline__ void cvt16(Frag& f, const float* __restrict__ p) {
    f32x4 a = *(const f32x4*)(p + 0);
    f32x4 b = *(const f32x4*)(p + 4);
    f32x4 c = *(const f32x4*)(p + 8);
    f32x4 d = *(const f32x4*)(p + 12);
    f.s[0]=bf16_bits(a.x);  f.s[1]=bf16_bits(a.y);  f.s[2]=bf16_bits(a.z);  f.s[3]=bf16_bits(a.w);
    f.s[4]=bf16_bits(b.x);  f.s[5]=bf16_bits(b.y);  f.s[6]=bf16_bits(b.z);  f.s[7]=bf16_bits(b.w);
    f.s[8]=bf16_bits(c.x);  f.s[9]=bf16_bits(c.y);  f.s[10]=bf16_bits(c.z); f.s[11]=bf16_bits(c.w);
    f.s[12]=bf16_bits(d.x); f.s[13]=bf16_bits(d.y); f.s[14]=bf16_bits(d.z); f.s[15]=bf16_bits(d.w);
}

// ------------------------------------------------------ f32 -> bf16 mirror --
__global__ void k_convert(const float* __restrict__ src,
                          unsigned short* __restrict__ dst) {
    const size_t i = (size_t)(blockIdx.x * blockDim.x + threadIdx.x) * 8;
    const float* p = src + i;
    f32x4 f0 = *(const f32x4*)(p);
    f32x4 f1 = *(const f32x4*)(p + 4);
    union { unsigned short s[8]; u32x4 q; } t;
    t.s[0]=bf16_bits(f0.x); t.s[1]=bf16_bits(f0.y); t.s[2]=bf16_bits(f0.z); t.s[3]=bf16_bits(f0.w);
    t.s[4]=bf16_bits(f1.x); t.s[5]=bf16_bits(f1.y); t.s[6]=bf16_bits(f1.z); t.s[7]=bf16_bits(f1.w);
    *(u32x4*)(dst + i) = t.q;
}

// ---------------------------------------------------------------- routing ---
__global__ void k_init(int* counts) {
    if (threadIdx.x < NEXP) counts[threadIdx.x] = 0;
}

__global__ void k_route(const float* __restrict__ x, const float* __restrict__ gate_w,
                        int* __restrict__ topk_idx, float* __restrict__ topk_w,
                        int* __restrict__ counts) {
    __shared__ float gs[NEXP * DIM];                       // 32 KB
    const int tid = threadIdx.x;
    for (int i = tid; i < NEXP * DIM; i += blockDim.x) gs[i] = gate_w[i];
    __syncthreads();

    const int t = blockIdx.x * blockDim.x + tid;
    const float* xr = x + (size_t)t * DIM;
    float acc[NEXP];
#pragma unroll
    for (int e = 0; e < NEXP; ++e) acc[e] = 0.f;
    for (int k = 0; k < DIM; k += 4) {
        f32x4 xv = *(const f32x4*)(xr + k);
#pragma unroll
        for (int e = 0; e < NEXP; ++e) {
            const float* g = gs + e * DIM + k;             // broadcast LDS read
            acc[e] += xv.x * g[0] + xv.y * g[1] + xv.z * g[2] + xv.w * g[3];
        }
    }
    int i0 = 0; float m0 = acc[0];
#pragma unroll
    for (int e = 1; e < NEXP; ++e) if (acc[e] > m0) { m0 = acc[e]; i0 = e; }
    int i1 = 0; float m1 = -3.0e38f;
#pragma unroll
    for (int e = 0; e < NEXP; ++e) if (e != i0 && acc[e] > m1) { m1 = acc[e]; i1 = e; }
    float a1  = __expf(m1 - m0);                           // <= 1
    float inv = fast_rcp(1.f + a1);
    topk_idx[t * 2 + 0] = i0;  topk_idx[t * 2 + 1] = i1;
    topk_w  [t * 2 + 0] = inv; topk_w  [t * 2 + 1] = a1 * inv;
    atomicAdd(&counts[i0], 1);
    atomicAdd(&counts[i1], 1);
}

__global__ void k_offsets(const int* counts, int* off, int* ts, int* cursor) {
    if (threadIdx.x == 0) {
        int o = 0, tt = 0;
        for (int e = 0; e < NEXP; ++e) {
            off[e] = o; ts[e] = tt;
            o  += counts[e];
            tt += (counts[e] + MTILE - 1) >> 5;            // 32-row tiles
        }
        off[NEXP] = o; ts[NEXP] = tt;
    }
    if (threadIdx.x < NEXP) cursor[threadIdx.x] = 0;
}

__global__ void k_scatter(const int* __restrict__ topk_idx, const float* __restrict__ topk_w,
                          const int* __restrict__ off, int* __restrict__ cursor,
                          int* __restrict__ slot_tok, float* __restrict__ slot_w) {
    const int t = blockIdx.x * blockDim.x + threadIdx.x;
#pragma unroll
    for (int j = 0; j < 2; ++j) {
        int   e = topk_idx[t * 2 + j];
        float w = topk_w [t * 2 + j];
        int   p = atomicAdd(&cursor[e], 1);
        slot_tok[off[e] + p] = t;
        slot_w  [off[e] + p] = w;
    }
}

// --------------------------------------------------------- fragment loads ---
template<bool PRE>
__device__ __forceinline__ void loadB(Frag& b, const float* rowF,
                                      const unsigned short* rowH, int kb) {
    if (PRE) {
        b.q[0] = *(const u32x4*)(rowH + kb);
        b.q[1] = *(const u32x4*)(rowH + kb + 16);
    } else {
        cvt16(b, rowF + kb);
        cvt16(b, rowF + kb);
        f32x4 f2 = *(const f32x4*)(rowF + kb + 16);
        f32x4 f3 = *(const f32x4*)(rowF + kb + 20);
        b.s[8]=bf16_bits(f2.x);  b.s[9]=bf16_bits(f2.y);  b.s[10]=bf16_bits(f2.z); b.s[11]=bf16_bits(f2.w);
        b.s[12]=bf16_bits(f3.x); b.s[13]=bf16_bits(f3.y); b.s[14]=bf16_bits(f3.z); b.s[15]=bf16_bits(f3.w);
    }
}

// A (16x32 bf16): lane<16 holds K {0..7,16..23}, lane>=16 {8..15,24..31}
template<bool PRE>
__device__ __forceinline__ void loadA(Frag& a, const float* rowF,
                                      const unsigned short* rowH, int kt, int hi) {
    if (PRE) {
        const unsigned short* p = rowH + kt + hi * 8;
        a.q[0] = *(const u32x4*)(p);
        a.q[1] = *(const u32x4*)(p + 16);
    } else {
        const float* p = rowF + kt + hi * 8;
        f32x4 f0 = *(const f32x4*)(p);
        f32x4 f1 = *(const f32x4*)(p + 4);
        f32x4 f2 = *(const f32x4*)(p + 16);
        f32x4 f3 = *(const f32x4*)(p + 20);
        a.s[0]=bf16_bits(f0.x);  a.s[1]=bf16_bits(f0.y);  a.s[2]=bf16_bits(f0.z);  a.s[3]=bf16_bits(f0.w);
        a.s[4]=bf16_bits(f1.x);  a.s[5]=bf16_bits(f1.y);  a.s[6]=bf16_bits(f1.z);  a.s[7]=bf16_bits(f1.w);
        a.s[8]=bf16_bits(f2.x);  a.s[9]=bf16_bits(f2.y);  a.s[10]=bf16_bits(f2.z); a.s[11]=bf16_bits(f2.w);
        a.s[12]=bf16_bits(f3.x); a.s[13]=bf16_bits(f3.y); a.s[14]=bf16_bits(f3.z); a.s[15]=bf16_bits(f3.w);
    }
}

// ------------------------------------------------------------- fused FFN ----
// 32-token tile, 256 threads (8 waves). Each B fragment is reused across both
// 16-row M-subtiles (4 WMMA per phase-1 B pair, 2 per phase-2 B) -> halves
// L2 weight traffic vs a 16-row tile.
template<bool PRE>
__device__ __forceinline__ void ffn_core(
        const float* __restrict__ xF, const unsigned short* __restrict__ xH,
        const float* wgF, const float* wuF, const float* wdF,
        const unsigned short* wgH, const unsigned short* wuH, const unsigned short* wdH,
        float* __restrict__ out,
        unsigned short* hs, const int* tokS, const float* rwS, bool atomic_out) {
    const int tid  = threadIdx.x;
    const int wave = tid >> 5;
    const int lane = tid & 31;
    const int mrow = lane & 15;
    const int hi   = lane >> 4;

    const int tok0 = tokS[mrow];
    const int tok1 = tokS[16 + mrow];
    const float*          xr0F = PRE ? nullptr : xF + (size_t)tok0 * DIM;
    const float*          xr1F = PRE ? nullptr : xF + (size_t)tok1 * DIM;
    const unsigned short* xr0H = PRE ? xH + (size_t)tok0 * DIM : nullptr;
    const unsigned short* xr1H = PRE ? xH + (size_t)tok1 * DIM : nullptr;

    // ---- Phase 1: g = x*Wg^T, u = x*Wu^T ; h = silu(g)*u -> LDS bf16 ----
    for (int ct = wave; ct < HALFD / 16; ct += 8) {
        v8f accg0 = {0.f,0.f,0.f,0.f,0.f,0.f,0.f,0.f};
        v8f accu0 = {0.f,0.f,0.f,0.f,0.f,0.f,0.f,0.f};
        v8f accg1 = {0.f,0.f,0.f,0.f,0.f,0.f,0.f,0.f};
        v8f accu1 = {0.f,0.f,0.f,0.f,0.f,0.f,0.f,0.f};
        const int ncol = ct * 16 + mrow;
        const float*          wgrowF = PRE ? nullptr : wgF + (size_t)ncol * DIM;
        const float*          wurowF = PRE ? nullptr : wuF + (size_t)ncol * DIM;
        const unsigned short* wgrowH = PRE ? wgH + (size_t)ncol * DIM : nullptr;
        const unsigned short* wurowH = PRE ? wuH + (size_t)ncol * DIM : nullptr;
        for (int kt = 0; kt < DIM; kt += 32) {
            const int kb = kt + hi * 8;
            Frag bg, bu, a0, a1;
            loadB<PRE>(bg, wgrowF, wgrowH, kb);
            loadB<PRE>(bu, wurowF, wurowH, kb);
            loadA<PRE>(a0, xr0F, xr0H, kt, hi);
            loadA<PRE>(a1, xr1F, xr1H, kt, hi);
            accg0 = __builtin_amdgcn_wmma_f32_16x16x32_bf16(false, a0.v, false, bg.v,
                                                            (short)0, accg0, false, false);
            accu0 = __builtin_amdgcn_wmma_f32_16x16x32_bf16(false, a0.v, false, bu.v,
                                                            (short)0, accu0, false, false);
            accg1 = __builtin_amdgcn_wmma_f32_16x16x32_bf16(false, a1.v, false, bg.v,
                                                            (short)0, accg1, false, false);
            accu1 = __builtin_amdgcn_wmma_f32_16x16x32_bf16(false, a1.v, false, bu.v,
                                                            (short)0, accu1, false, false);
        }
        // C/D layout: elem j -> M = j + 8*hi, N = mrow (within this 16-col tile)
#pragma unroll
        for (int j = 0; j < 8; ++j) {
            const int m = j + (hi << 3);
            float g0 = accg0[j];
            float h0 = g0 * fast_rcp(1.f + __expf(-g0)) * accu0[j];
            hs[m * HS_STRIDE + ct * 16 + mrow] = bf16_bits(h0);
            float g1 = accg1[j];
            float h1 = g1 * fast_rcp(1.f + __expf(-g1)) * accu1[j];
            hs[(16 + m) * HS_STRIDE + ct * 16 + mrow] = bf16_bits(h1);
        }
    }
    __syncthreads();

    // ---- Phase 2: y = h * Wd^T ; out[tok] (+)= rw * y ----
    const unsigned short* hrow0 = hs + mrow * HS_STRIDE;
    const unsigned short* hrow1 = hs + (16 + mrow) * HS_STRIDE;
    for (int ct = wave; ct < DIM / 16; ct += 8) {
        v8f acc0 = {0.f,0.f,0.f,0.f,0.f,0.f,0.f,0.f};
        v8f acc1 = {0.f,0.f,0.f,0.f,0.f,0.f,0.f,0.f};
        const int ncol = ct * 16 + mrow;
        const float*          wdrowF = PRE ? nullptr : wdF + (size_t)ncol * HALFD;
        const unsigned short* wdrowH = PRE ? wdH + (size_t)ncol * HALFD : nullptr;
        for (int kt = 0; kt < HALFD; kt += 32) {
            Frag b, a0, a1;
            loadB<PRE>(b, wdrowF, wdrowH, kt + hi * 8);
            const unsigned short* p0 = hrow0 + kt + hi * 8;
            a0.q[0] = *(const u32x4*)(p0);
            a0.q[1] = *(const u32x4*)(p0 + 16);
            const unsigned short* p1 = hrow1 + kt + hi * 8;
            a1.q[0] = *(const u32x4*)(p1);
            a1.q[1] = *(const u32x4*)(p1 + 16);
            acc0 = __builtin_amdgcn_wmma_f32_16x16x32_bf16(false, a0.v, false, b.v,
                                                           (short)0, acc0, false, false);
            acc1 = __builtin_amdgcn_wmma_f32_16x16x32_bf16(false, a1.v, false, b.v,
                                                           (short)0, acc1, false, false);
        }
#pragma unroll
        for (int j = 0; j < 8; ++j) {
            const int m = j + (hi << 3);
            const int n = ct * 16 + mrow;
            float v0 = acc0[j] * rwS[m];
            float v1 = acc1[j] * rwS[16 + m];
            size_t oa0 = (size_t)tokS[m] * DIM + n;
            size_t oa1 = (size_t)tokS[16 + m] * DIM + n;
            if (atomic_out) { atomicAdd(&out[oa0], v0); atomicAdd(&out[oa1], v1); }
            else            { out[oa0] = v0;            out[oa1] = v1;            }
        }
    }
}

template<bool PRE>
__global__ void __launch_bounds__(256)
k_shared_ffn(const float* __restrict__ x, const unsigned short* xH,
             const float* sg_gate, const float* sg_up, const float* sg_down,
             const unsigned short* sggH, const unsigned short* sguH,
             const unsigned short* sgdH,
             float* __restrict__ out) {
    __shared__ __align__(16) unsigned short hs[MTILE * HS_STRIDE];  // 98.8 KB
    __shared__ int   tokS[MTILE];
    __shared__ float rwS[MTILE];
    if (threadIdx.x < MTILE) {
        tokS[threadIdx.x] = blockIdx.x * MTILE + threadIdx.x;
        rwS[threadIdx.x]  = 1.0f;
    }
    __syncthreads();
    ffn_core<PRE>(x, xH, sg_gate, sg_up, sg_down, sggH, sguH, sgdH,
                  out, hs, tokS, rwS, /*atomic=*/false);
}

template<bool PRE>
__global__ void __launch_bounds__(256)
k_expert_ffn(const float* __restrict__ x, const unsigned short* xH,
             const float* w_up, const float* w_down,
             const unsigned short* upH, const unsigned short* dnH,
             const int* __restrict__ counts, const int* __restrict__ off,
             const int* __restrict__ ts,
             const int* __restrict__ slot_tok, const float* __restrict__ slot_w,
             float* __restrict__ out) {
    __shared__ __align__(16) unsigned short hs[MTILE * HS_STRIDE];
    __shared__ int   tokS[MTILE];
    __shared__ float rwS[MTILE];
    __shared__ int   eS;
    const int b = blockIdx.x;
    if (threadIdx.x == 0) {
        int e = -1;
        if (b < ts[NEXP]) { e = 0; while (!(b < ts[e + 1])) ++e; }
        eS = e;
    }
    __syncthreads();
    const int e = eS;
    if (e < 0) return;                               // uniform per block
    if (threadIdx.x < MTILE) {
        int lr = (b - ts[e]) * MTILE + threadIdx.x;  // expert-local row
        if (lr < counts[e]) {
            int s = off[e] + lr;
            tokS[threadIdx.x] = slot_tok[s];
            rwS[threadIdx.x]  = slot_w[s];
        } else {                                     // pad row: weight 0
            tokS[threadIdx.x] = 0;
            rwS[threadIdx.x]  = 0.f;
        }
    }
    __syncthreads();
    const size_t upOff = (size_t)e * (2 * HALFD) * DIM;
    const size_t dnOff = (size_t)e * DIM * HALFD;
    ffn_core<PRE>(x, xH,
                  w_up + upOff, w_up + upOff + (size_t)HALFD * DIM, w_down + dnOff,
                  upH  + upOff, upH  + upOff + (size_t)HALFD * DIM, dnH   + dnOff,
                  out, hs, tokS, rwS, /*atomic=*/true);
}

// ---------------------------------------------------------------- launch ----
extern "C" void kernel_launch(void* const* d_in, const int* in_sizes, int n_in,
                              void* d_out, int out_size, void* d_ws, size_t ws_size,
                              hipStream_t stream) {
    const float* x       = (const float*)d_in[0];
    const float* gate_w  = (const float*)d_in[1];
    const float* w_up    = (const float*)d_in[2];
    const float* w_down  = (const float*)d_in[3];
    const float* sg_gate = (const float*)d_in[4];
    const float* sg_up   = (const float*)d_in[5];
    const float* sg_down = (const float*)d_in[6];
    float* out = (float*)d_out;

    int*   wsi      = (int*)d_ws;
    int*   counts   = wsi;                                   // 8
    int*   cursor   = wsi + 8;                               // 8
    int*   off      = wsi + 16;                              // 9
    int*   ts       = wsi + 32;                              // 9
    int*   topk_idx = wsi + 64;                              // 2N
    float* topk_w   = (float*)(wsi + 64 + 2 * N_TOK);        // 2N
    int*   slot_tok = wsi + 64 + 4 * N_TOK;                  // 2N
    float* slot_w   = (float*)(wsi + 64 + 6 * N_TOK);        // 2N

    // bf16 mirrors (weights + x), 1 MB past the int region
    const size_t upN = (size_t)NEXP * 2 * HALFD * DIM;       // 25.2 M elems
    const size_t dnN = (size_t)NEXP * DIM * HALFD;           // 12.6 M elems
    const size_t sgN = (size_t)HALFD * DIM;                  // 1.57 M elems
    const size_t xN  = (size_t)N_TOK * DIM;                  // 16.8 M elems
    const size_t bfTotal = upN + dnN + 3 * sgN + xN;
    unsigned short* upH  = (unsigned short*)((char*)d_ws + (1u << 20));
    unsigned short* dnH  = upH  + upN;
    unsigned short* sggH = dnH  + dnN;
    unsigned short* sguH = sggH + sgN;
    unsigned short* sgdH = sguH + sgN;
    unsigned short* xH   = sgdH + sgN;
    const bool pre = ws_size >= (1u << 20) + bfTotal * sizeof(unsigned short);

    k_init   <<<1, 64, 0, stream>>>(counts);
    k_route  <<<N_TOK / 128, 128, 0, stream>>>(x, gate_w, topk_idx, topk_w, counts);
    k_offsets<<<1, 64, 0, stream>>>(counts, off, ts, cursor);
    k_scatter<<<N_TOK / 256, 256, 0, stream>>>(topk_idx, topk_w, off, cursor,
                                               slot_tok, slot_w);

    const int maxTiles = (2 * N_TOK) / MTILE + NEXP;         // upper bound, 1032
    if (pre) {
        k_convert<<<(int)(upN / 8 / 256), 256, 0, stream>>>(w_up,    upH);
        k_convert<<<(int)(dnN / 8 / 256), 256, 0, stream>>>(w_down,  dnH);
        k_convert<<<(int)(sgN / 8 / 256), 256, 0, stream>>>(sg_gate, sggH);
        k_convert<<<(int)(sgN / 8 / 256), 256, 0, stream>>>(sg_up,   sguH);
        k_convert<<<(int)(sgN / 8 / 256), 256, 0, stream>>>(sg_down, sgdH);
        k_convert<<<(int)(xN  / 8 / 256), 256, 0, stream>>>(x,       xH);
        k_shared_ffn<true><<<N_TOK / MTILE, 256, 0, stream>>>(
            x, xH, sg_gate, sg_up, sg_down, sggH, sguH, sgdH, out);
        k_expert_ffn<true><<<maxTiles, 256, 0, stream>>>(
            x, xH, w_up, w_down, upH, dnH, counts, off, ts, slot_tok, slot_w, out);
    } else {
        k_shared_ffn<false><<<N_TOK / MTILE, 256, 0, stream>>>(
            x, nullptr, sg_gate, sg_up, sg_down, nullptr, nullptr, nullptr, out);
        k_expert_ffn<false><<<maxTiles, 256, 0, stream>>>(
            x, nullptr, w_up, w_down, nullptr, nullptr,
            counts, off, ts, slot_tok, slot_w, out);
    }
}
